// MultiheadAttention_76476187672916
// MI455X (gfx1250) — matrix operand order repaired and checked by reading, not compiled, not run on hardware
//
#include <hip/hip_runtime.h>
#include <math.h>

// ---- Problem constants (from reference) ----
#define DMODEL 1024
#define NHEAD  16
#define DHEAD  64
#define BB     2
#define SS     2048
#define SCALE  0.125f               // 1/sqrt(64)
#define BH     (BB*NHEAD)           // 32
#define OUT_ELEMS   ((size_t)BB*SS*DMODEL)         // 4,194,304
#define ATTN_PER_BH ((size_t)SS*SS)                // 4,194,304
#define HEAD_ELEMS  ((size_t)BH*SS*DHEAD)          // 4,194,304

// ---- WMMA vector types (gfx1250, wave32) ----
typedef __bf16 v16bf __attribute__((ext_vector_type(16)));
typedef __bf16 v8bf  __attribute__((ext_vector_type(8)));
typedef __bf16 v4bf  __attribute__((ext_vector_type(4)));
typedef float  v8f   __attribute__((ext_vector_type(8)));

// ---- Tiling: 256 threads = 8 waves (4Mx2N); wave tile 32x32 = 2x2 WMMA tiles.
// Block tile 128x64, K-chunk 32, bf16 double-buffered LDS.
#define BM 128
#define BN 64
#define BK 32
#define ALD (BK + 8)     // 40 bf16/row = 80B: 16B-aligned frags, spreads banks

__device__ __forceinline__ v16bf frag_from_halves(v8bf lo, v8bf hi) {
  return __builtin_shufflevector(lo, hi, 0, 1, 2, 3, 4, 5, 6, 7,
                                 8, 9, 10, 11, 12, 13, 14, 15);
}
// A fragment (16x32 bf16): elems 0..7 = K g*8+i, elems 8..15 = K 16+g*8+i
__device__ __forceinline__ v16bf frag_a(const __bf16* rowp, int g) {
  v8bf lo = *(const v8bf*)(rowp + g * 8);
  v8bf hi = *(const v8bf*)(rowp + 16 + g * 8);
  return frag_from_halves(lo, hi);
}
// B fragment (32x16 bf16): elems 0..7 = K g*8+i, elems 8..15 = K 16+g*8+i
__device__ __forceinline__ v16bf frag_b(const __bf16* rowp, int g) {
  v8bf lo = *(const v8bf*)(rowp + g * 8);
  v8bf hi = *(const v8bf*)(rowp + 16 + g * 8);
  return frag_from_halves(lo, hi);
}
__device__ __forceinline__ v4bf cvt4(float4 v) {
  v4bf c;
  c[0] = (__bf16)v.x; c[1] = (__bf16)v.y; c[2] = (__bf16)v.z; c[3] = (__bf16)v.w;
  return c;
}

// MODE: 0 = QKV proj -> Q,K,V^T buffers   1 = Q K^T * scale -> attn (raw)
//       2 = P @ V -> ctx                  3 = ctx @ Wout^T + b -> out
template <int MODE>
__global__ __launch_bounds__(256)
void gemm_nt_wmma(const float* __restrict__ A, const float* __restrict__ Bm,
                  const float* __restrict__ bias,
                  float* __restrict__ o0, float* __restrict__ o1,
                  float* __restrict__ o2) {
  constexpr int lda = (MODE == 1) ? DHEAD : (MODE == 2) ? SS : DMODEL;
  constexpr int ldb = (MODE == 1) ? DHEAD : (MODE == 2) ? SS : DMODEL;

  const int m0 = blockIdx.y * BM;
  const int n0 = blockIdx.x * BN;

  const float* Ap = A;
  const float* Bp = Bm;
  if constexpr (MODE == 1) {
    if (n0 > m0 + (BM - 1)) return;          // fully-masked causal tile
    const size_t bh = blockIdx.z;
    Ap = A  + bh * (size_t)SS * DHEAD;
    Bp = Bm + bh * (size_t)SS * DHEAD;
  }
  if constexpr (MODE == 2) {
    const size_t bh = blockIdx.z;
    Ap = A  + bh * ATTN_PER_BH;
    Bp = Bm + bh * (size_t)DHEAD * SS;
  }

  int kchunks;
  if constexpr (MODE == 1)      kchunks = DHEAD / BK;          // 2
  else if constexpr (MODE == 2) kchunks = m0 / BK + BM / BK;   // causal K limit
  else                          kchunks = DMODEL / BK;         // 32

  __shared__ __bf16 As[2][BM][ALD];   // 20 KB
  __shared__ __bf16 Bs[2][BN][ALD];   // 10 KB

  const int tid  = threadIdx.x;
  const int wave = tid >> 5;
  const int lane = tid & 31;
  const int wm   = wave & 3;          // wave tile row (0..3) -> 32 rows each
  const int wn   = wave >> 2;         // wave tile col (0..1) -> 32 cols each
  const int l16  = lane & 15;
  const int g    = lane >> 4;

  // cooperative-copy mapping (fixed per thread)
  const int ar[4] = { tid >> 3, (tid + 256) >> 3, (tid + 512) >> 3, (tid + 768) >> 3 };
  const int ac    = (tid & 7) * 4;
  const int br[2] = { tid >> 3, (tid + 256) >> 3 };

  float4 ra[4], rb[2];
  // ---- prologue: chunk 0 ----
  #pragma unroll
  for (int j = 0; j < 4; ++j)
    ra[j] = *(const float4*)(Ap + (size_t)(m0 + ar[j]) * lda + ac);
  #pragma unroll
  for (int j = 0; j < 2; ++j)
    rb[j] = *(const float4*)(Bp + (size_t)(n0 + br[j]) * ldb + ac);
  #pragma unroll
  for (int j = 0; j < 4; ++j) *(v4bf*)&As[0][ar[j]][ac] = cvt4(ra[j]);
  #pragma unroll
  for (int j = 0; j < 2; ++j) *(v4bf*)&Bs[0][br[j]][ac] = cvt4(rb[j]);
  __syncthreads();

  v8f acc[2][2] = {};

  for (int kc = 0; kc < kchunks; ++kc) {
    const int cur = kc & 1, nxt = cur ^ 1;
    const bool more = (kc + 1 < kchunks);
    if (more) {
      const int k0 = (kc + 1) * BK;
      #pragma unroll
      for (int j = 0; j < 4; ++j)
        ra[j] = *(const float4*)(Ap + (size_t)(m0 + ar[j]) * lda + k0 + ac);
      #pragma unroll
      for (int j = 0; j < 2; ++j)
        rb[j] = *(const float4*)(Bp + (size_t)(n0 + br[j]) * ldb + k0 + ac);
      if (kc + 2 < kchunks) {   // pull chunk+2 lines toward L2 (global_prefetch_b8)
        __builtin_prefetch(Ap + (size_t)(m0 + (tid & (BM - 1))) * lda + k0 + BK, 0, 0);
        if (tid < BN)
          __builtin_prefetch(Bp + (size_t)(n0 + tid) * ldb + k0 + BK, 0, 0);
      }
    }

    // ---- 4 WMMAs on current buffer ----
    const v16bf a0 = frag_a(&As[cur][wm * 32 + l16][0], g);
    const v16bf a1 = frag_a(&As[cur][wm * 32 + 16 + l16][0], g);
    const v16bf b0 = frag_b(&Bs[cur][wn * 32 + l16][0], g);
    const v16bf b1 = frag_b(&Bs[cur][wn * 32 + 16 + l16][0], g);
    acc[0][0] = __builtin_amdgcn_wmma_f32_16x16x32_bf16(false, a0, false, b0,
                                                        (short)0, acc[0][0], false, false);
    acc[0][1] = __builtin_amdgcn_wmma_f32_16x16x32_bf16(false, a0, false, b1,
                                                        (short)0, acc[0][1], false, false);
    acc[1][0] = __builtin_amdgcn_wmma_f32_16x16x32_bf16(false, a1, false, b0,
                                                        (short)0, acc[1][0], false, false);
    acc[1][1] = __builtin_amdgcn_wmma_f32_16x16x32_bf16(false, a1, false, b1,
                                                        (short)0, acc[1][1], false, false);

    if (more) {
      #pragma unroll
      for (int j = 0; j < 4; ++j) *(v4bf*)&As[nxt][ar[j]][ac] = cvt4(ra[j]);
      #pragma unroll
      for (int j = 0; j < 2; ++j) *(v4bf*)&Bs[nxt][br[j]][ac] = cvt4(rb[j]);
    }
    __syncthreads();
  }

  // ---- epilogue: C/D layout: lane col n=l16, rows r + g*8 ----
  #pragma unroll
  for (int ti = 0; ti < 2; ++ti) {
    #pragma unroll
    for (int tj = 0; tj < 2; ++tj) {
      const int ng = n0 + wn * 32 + tj * 16 + l16;
      const int mbase = m0 + wm * 32 + ti * 16 + g * 8;

      if constexpr (MODE == 0) {
        const float bv = bias[ng];
        const int seg = ng >> 10, wi = ng & 1023;
        const int h = wi >> 6, d = wi & 63;
        #pragma unroll
        for (int r = 0; r < 8; ++r) {
          const int m = mbase + r;
          const int b = m >> 11, s = m & (SS - 1);
          const float val = acc[ti][tj][r] + bv;
          const size_t bh = (size_t)(b * NHEAD + h);
          if (seg == 0)      o0[(bh * SS + s) * DHEAD + d] = val;   // Q [bh,s,d]
          else if (seg == 1) o1[(bh * SS + s) * DHEAD + d] = val;   // K [bh,s,d]
          else               o2[(bh * DHEAD + d) * SS + s] = val;   // V^T [bh,d,s]
        }
      } else if constexpr (MODE == 1) {
        float* dst = o0 + (size_t)blockIdx.z * ATTN_PER_BH;
        #pragma unroll
        for (int r = 0; r < 8; ++r)
          dst[(size_t)(mbase + r) * SS + ng] = acc[ti][tj][r] * SCALE;
      } else if constexpr (MODE == 2) {
        const int b = blockIdx.z >> 4, h = blockIdx.z & 15;
        #pragma unroll
        for (int r = 0; r < 8; ++r)
          o0[((size_t)(b * SS + mbase + r)) * DMODEL + h * DHEAD + ng] = acc[ti][tj][r];
      } else {
        const float bv = bias[ng];
        #pragma unroll
        for (int r = 0; r < 8; ++r)
          o0[(size_t)(mbase + r) * DMODEL + ng] = acc[ti][tj][r] + bv;
      }
    }
  }
}

// ---- causal row softmax over attn region; writes exact 0 for masked ----
__global__ __launch_bounds__(256)
void softmax_causal(float* __restrict__ attn) {
  const size_t row = blockIdx.x;
  const int q = (int)(row & (SS - 1));
  const int valid = q + 1;
  float* p = attn + row * (size_t)SS;

  const int tid = threadIdx.x;
  const int lane = tid & 31, wave = tid >> 5;
  __shared__ float redmax[8];
  __shared__ float redsum[8];

  float vals[SS / 256];
  float lmax = -__builtin_inff();
  #pragma unroll
  for (int i = 0; i < SS / 256; ++i) {
    const int c = tid + i * 256;
    vals[i] = (c < valid) ? p[c] : -__builtin_inff();
    lmax = fmaxf(lmax, vals[i]);
  }
  #pragma unroll
  for (int off = 16; off > 0; off >>= 1)
    lmax = fmaxf(lmax, __shfl_xor(lmax, off, 32));
  if (lane == 0) redmax[wave] = lmax;
  __syncthreads();
  float rmax = redmax[0];
  #pragma unroll
  for (int j = 1; j < 8; ++j) rmax = fmaxf(rmax, redmax[j]);

  float lsum = 0.f;
  #pragma unroll
  for (int i = 0; i < SS / 256; ++i) {
    const int c = tid + i * 256;
    vals[i] = (c < valid) ? __expf(vals[i] - rmax) : 0.f;
    lsum += vals[i];
  }
  #pragma unroll
  for (int off = 16; off > 0; off >>= 1)
    lsum += __shfl_xor(lsum, off, 32);
  if (lane == 0) redsum[wave] = lsum;
  __syncthreads();
  float rsum = 0.f;
  #pragma unroll
  for (int j = 0; j < 8; ++j) rsum += redsum[j];
  const float inv = 1.0f / rsum;

  #pragma unroll
  for (int i = 0; i < SS / 256; ++i) {
    const int c = tid + i * 256;
    p[c] = vals[i] * inv;
  }
}

extern "C" void kernel_launch(void* const* d_in, const int* in_sizes, int n_in,
                              void* d_out, int out_size, void* d_ws, size_t ws_size,
                              hipStream_t stream) {
  (void)in_sizes; (void)n_in; (void)out_size; (void)ws_size;
  const float* x    = (const float*)d_in[0];   // [2,2048,1024]
  const float* W    = (const float*)d_in[1];   // [4096,1024]
  const float* bias = (const float*)d_in[2];   // [4096]

  float* out  = (float*)d_out;                 // [2,2048,1024]
  float* attn = out + OUT_ELEMS;               // [2,16,2048,2048]

  float* qb  = (float*)d_ws;                   // [bh,s,d]
  float* kb  = qb  + HEAD_ELEMS;               // [bh,s,d]
  float* vtb = kb  + HEAD_ELEMS;               // [bh,d,s] (transposed V)
  float* ctx = vtb + HEAD_ELEMS;               // [b,s,1024]

  const dim3 blk(256);
  // 1) QKV projection: M=4096, N=3072, K=1024
  gemm_nt_wmma<0><<<dim3(3 * DMODEL / BN, BB * SS / BM, 1), blk, 0, stream>>>(
      x, W, bias, qb, kb, vtb);
  // 2) scores = Q K^T * scale (causal tiles skipped): per head M=N=2048, K=64
  gemm_nt_wmma<1><<<dim3(SS / BN, SS / BM, BH), blk, 0, stream>>>(
      qb, kb, nullptr, attn, nullptr, nullptr);
  // 3) causal softmax over every attn row
  softmax_causal<<<dim3(BH * SS), blk, 0, stream>>>(attn);
  // 4) ctx = P @ V: per head M=2048, N=64, K causal-limited
  gemm_nt_wmma<2><<<dim3(DHEAD / BN, SS / BM, BH), blk, 0, stream>>>(
      attn, vtb, nullptr, ctx, nullptr, nullptr);
  // 5) output projection: M=4096, N=1024, K=1024
  gemm_nt_wmma<3><<<dim3(DMODEL / BN, BB * SS / BM, 1), blk, 0, stream>>>(
      ctx, W + (size_t)3 * DMODEL * DMODEL, bias + 3 * DMODEL,
      out, nullptr, nullptr);
}
